// SlimNNAttentionHead_13073880449824
// MI455X (gfx1250) — compile-verified, matrix operand order checked by reading, not run on hardware
//
#include <hip/hip_runtime.h>
#include <hip/hip_bf16.h>

typedef __attribute__((ext_vector_type(16))) __bf16 v16bf;
typedef __attribute__((ext_vector_type(8)))  float  v8f;

#define BATCH 2
#define T 768
#define C 64
#define HS 64
#define NROW (BATCH * T)        // 1536
#define TILES_PER_B (T / 16)    // 48

// workspace byte offsets
#define OFF_PW1B   0u           // 64*64 bf16
#define OFF_PW2B   8192u        // 64*64 bf16
#define OFF_AW1B   16384u       // 256*64 bf16
#define OFF_VWB    49152u       // 64*64 bf16
#define OFF_WVEC   57344u       // 64 f32
#define OFF_SBIAS  57600u       // 1 f32 (padded)
#define OFF_HK     57856u       // 1536*64 f32
#define OFF_HQ     451072u      // 1536*64 f32
#define OFF_V      844288u      // 1536*64 f32

__device__ __forceinline__ v8f wmma_bf16(v16bf a, v16bf b, v8f c) {
  return __builtin_amdgcn_wmma_f32_16x16x32_bf16(false, a, false, b, (short)0, c,
                                                 false, false);
}

// A fragment (16x32 bf16) from a padded f32 LDS tile (row stride ldm floats).
// Lane L: row m = L%16 ; element e holds K = kbase + (e>=8?16:0) + 8*(L/16) + e%8
__device__ __forceinline__ v16bf load_a_lds(const float* tile, int lane, int kbase,
                                            int ldm) {
  v16bf a;
  int m = lane & 15;
  int kh = (lane >> 4) << 3;
#pragma unroll
  for (int e = 0; e < 16; ++e) {
    int k = kbase + ((e & 8) << 1) + kh + (e & 7);
    a[e] = (__bf16)tile[m * ldm + k];
  }
  return a;
}

// B fragment (32x16 bf16) from a row-major bf16 weight [K x C].
// Lane L: col n = nbase + L%16 ; element e holds K = kbase + (e>=8?16:0) + 8*(L/16) + e%8
__device__ __forceinline__ v16bf load_b_w(const __bf16* W, int lane, int kbase,
                                          int nbase) {
  v16bf b;
  int n = nbase + (lane & 15);
  int kh = (lane >> 4) << 3;
#pragma unroll
  for (int e = 0; e < 16; ++e) {
    int k = kbase + ((e & 8) << 1) + kh + (e & 7);
    b[e] = W[k * C + n];
  }
  return b;
}

// Contiguous 8KB global -> LDS async copy (32 lanes x 16 x b128).
// Tracked by ASYNCcnt; caller synchronizes with s_wait_asynccnt.
__device__ __forceinline__ void async_copy_8k(unsigned lds_base, const float* gsrc,
                                              int lane) {
  unsigned dst = lds_base + (unsigned)(lane * 16);
  unsigned long long src =
      (unsigned long long)(uintptr_t)gsrc + (unsigned long long)(lane * 16);
#pragma unroll
  for (int i = 0; i < 16; ++i) {
    asm volatile("global_load_async_to_lds_b128 %0, %1, off" ::"v"(dst), "v"(src)
                 : "memory");
    dst += 512u;
    src += 512ull;
  }
}

// ---------------- kernel 0: weight prep + algebraic folding ----------------
__global__ __launch_bounds__(256) void prep_kernel(
    const float* pw1, const float* pw2, const float* aw1, const float* vw,
    const float* aw2, const float* sw, const float* ab2, const float* sb,
    __bf16* pw1b, __bf16* pw2b, __bf16* aw1b, __bf16* vwb, float* wvec,
    float* sbias) {
  int tid = threadIdx.x;
  for (int i = tid; i < C * C; i += 256) pw1b[i] = (__bf16)pw1[i];
  for (int i = tid; i < C * C; i += 256) pw2b[i] = (__bf16)pw2[i];
  for (int i = tid; i < 4 * C * C; i += 256) aw1b[i] = (__bf16)aw1[i];
  for (int i = tid; i < C * HS; i += 256) vwb[i] = (__bf16)vw[i];
  if (tid < C) {
    float s = 0.f;
    for (int n = 0; n < C; ++n) s += aw2[tid * C + n] * sw[n];
    wvec[tid] = s;
  }
  if (tid == 0) {
    float s = sb[0];
    for (int n = 0; n < C; ++n) s += ab2[n] * sw[n];
    sbias[0] = s;
  }
}

// ------- kernel 1: pe feed-forward + hk/hq/v projections (WMMA) per 16 rows -------
__global__ __launch_bounds__(32) void proj_kernel(
    const float* x, const float* pe_in, const float* pb1, const float* pb2,
    const float* ab1, const __bf16* pw1b, const __bf16* pw2b, const __bf16* aw1b,
    const __bf16* vwb, float* g_hk, float* g_hq, float* g_v) {
  __shared__ float S[16 * 65];
  __shared__ float X[16 * 65];
  __shared__ float T1[16 * 65];
  __shared__ float PE[16 * 65];

  int lane = threadIdx.x;
  int row0 = blockIdx.x * 16;  // flattened over B*T

  for (int idx = lane; idx < 16 * C; idx += 32) {
    int r = idx >> 6, c = idx & 63;
    S[r * 65 + c] = pe_in[(row0 + r) * C + c];
    X[r * 65 + c] = x[(row0 + r) * C + c];
  }
  __syncthreads();

  int m = lane & 15;
  int kh = (lane >> 4) << 3;

  // t1 = relu(S @ pw1 + pb1)
  v16bf aS0 = load_a_lds(S, lane, 0, 65);
  v16bf aS1 = load_a_lds(S, lane, 32, 65);
#pragma unroll
  for (int nc = 0; nc < 4; ++nc) {
    v8f c = {};
    c = wmma_bf16(aS0, load_b_w(pw1b, lane, 0, nc * 16), c);
    c = wmma_bf16(aS1, load_b_w(pw1b, lane, 32, nc * 16), c);
    int n = nc * 16 + m;
#pragma unroll
    for (int r = 0; r < 8; ++r) T1[(r + kh) * 65 + n] = fmaxf(c[r] + pb1[n], 0.f);
  }
  __syncthreads();

  // pe = t1 @ pw2 + pb2
  v16bf aT0 = load_a_lds(T1, lane, 0, 65);
  v16bf aT1 = load_a_lds(T1, lane, 32, 65);
#pragma unroll
  for (int nc = 0; nc < 4; ++nc) {
    v8f c = {};
    c = wmma_bf16(aT0, load_b_w(pw2b, lane, 0, nc * 16), c);
    c = wmma_bf16(aT1, load_b_w(pw2b, lane, 32, nc * 16), c);
    int n = nc * 16 + m;
#pragma unroll
    for (int r = 0; r < 8; ++r) PE[(r + kh) * 65 + n] = c[r] + pb2[n];
  }
  __syncthreads();

  v16bf aP0 = load_a_lds(PE, lane, 0, 65);
  v16bf aP1 = load_a_lds(PE, lane, 32, 65);
  v16bf aX0 = load_a_lds(X, lane, 0, 65);
  v16bf aX1 = load_a_lds(X, lane, 32, 65);

  // hk = pe @ aw1[0:64] + x @ aw1[64:128]
#pragma unroll
  for (int nc = 0; nc < 4; ++nc) {
    v8f c = {};
    c = wmma_bf16(aP0, load_b_w(aw1b, lane, 0, nc * 16), c);
    c = wmma_bf16(aP1, load_b_w(aw1b, lane, 32, nc * 16), c);
    c = wmma_bf16(aX0, load_b_w(aw1b, lane, 64, nc * 16), c);
    c = wmma_bf16(aX1, load_b_w(aw1b, lane, 96, nc * 16), c);
    int n = nc * 16 + m;
#pragma unroll
    for (int r = 0; r < 8; ++r) g_hk[(row0 + r + kh) * C + n] = c[r];
  }
  // hq = pe @ aw1[128:192] + x @ aw1[192:256] + ab1   (ab1 folded here)
#pragma unroll
  for (int nc = 0; nc < 4; ++nc) {
    v8f c = {};
    c = wmma_bf16(aP0, load_b_w(aw1b, lane, 128, nc * 16), c);
    c = wmma_bf16(aP1, load_b_w(aw1b, lane, 160, nc * 16), c);
    c = wmma_bf16(aX0, load_b_w(aw1b, lane, 192, nc * 16), c);
    c = wmma_bf16(aX1, load_b_w(aw1b, lane, 224, nc * 16), c);
    int n = nc * 16 + m;
#pragma unroll
    for (int r = 0; r < 8; ++r) g_hq[(row0 + r + kh) * C + n] = c[r] + ab1[n];
  }
  // v = x @ vw
#pragma unroll
  for (int nc = 0; nc < 4; ++nc) {
    v8f c = {};
    c = wmma_bf16(aX0, load_b_w(vwb, lane, 0, nc * 16), c);
    c = wmma_bf16(aX1, load_b_w(vwb, lane, 32, nc * 16), c);
    int n = nc * 16 + m;
#pragma unroll
    for (int r = 0; r < 8; ++r) g_v[(row0 + r + kh) * C + n] = c[r];
  }
}

// ------- kernel 2: causal flash softmax + P@V via WMMA, 16 query rows / wave -------
// hk/v chunks staged with double-buffered GLOBAL_LOAD_ASYNC_TO_LDS_B128 (ASYNCcnt).
__global__ __launch_bounds__(32) void attn_kernel(const float* g_hq, const float* g_hk,
                                                  const float* g_v, const float* wvec,
                                                  const float* sbias_p, float* out) {
  __shared__ float HQ[16 * 65];      // padded: per-row reads indexed by lane
  __shared__ float HK[2][32 * 64];   // ping-pong async buffers (broadcast reads)
  __shared__ float V[2][32 * 64];
  __shared__ float W[C];

  int lane = threadIdx.x;
  int b = blockIdx.x / TILES_PER_B;
  int it = blockIdx.x % TILES_PER_B;
  int rowbase = b * T + it * 16;
  int nchunk = (it * 16 + 16 + 31) >> 5;  // 32-wide j chunks covering [0, it*16+16)

  unsigned hk_lds[2] = {(unsigned)(uintptr_t)&HK[0][0], (unsigned)(uintptr_t)&HK[1][0]};
  unsigned v_lds[2]  = {(unsigned)(uintptr_t)&V[0][0],  (unsigned)(uintptr_t)&V[1][0]};

  // kick off chunk 0 DMA, then overlap with the synchronous HQ/W staging
  async_copy_8k(hk_lds[0], g_hk + (size_t)(b * T) * C, lane);
  async_copy_8k(v_lds[0],  g_v  + (size_t)(b * T) * C, lane);

  for (int idx = lane; idx < 16 * C; idx += 32) {
    int r = idx >> 6, c = idx & 63;
    HQ[r * 65 + c] = g_hq[(rowbase + r) * C + c];
  }
  for (int idx = lane; idx < C; idx += 32) W[idx] = wvec[idx];
  float sbias = sbias_p[0];

  int m = lane & 15;
  int kh = (lane >> 4) << 3;
  int ilocal = it * 16 + m;  // this lane's query row (within batch)

  int jl[16];  // A-fragment K slots == local j indices this lane owns
#pragma unroll
  for (int e = 0; e < 16; ++e) jl[e] = ((e & 8) << 1) + kh + (e & 7);

  v8f acc[4] = {v8f{}, v8f{}, v8f{}, v8f{}};
  float mrow = -__builtin_inff();
  float lsum = 0.f;

  for (int ch = 0; ch < nchunk; ++ch) {
    int buf = ch & 1;
    if (ch + 1 < nchunk) {
      // WAR guard: all ds reads of the buffer we are about to overwrite are done
      asm volatile("s_wait_dscnt 0x0" ::: "memory");
      int jn = (ch + 1) << 5;
      async_copy_8k(hk_lds[buf ^ 1], g_hk + (size_t)(b * T + jn) * C, lane);
      async_copy_8k(v_lds[buf ^ 1],  g_v  + (size_t)(b * T + jn) * C, lane);
      // wait until only the 32 next-chunk ops remain in flight -> current landed
      asm volatile("s_wait_asynccnt 0x20" ::: "memory");
    } else {
      asm volatile("s_wait_asynccnt 0x0" ::: "memory");
    }
    __syncthreads();

    int jbase = ch << 5;
    const float* HKc = &HK[buf][0];
    const float* Vc = &V[buf][0];

    // scores for this lane's 16 (i=m, j=jbase+jl[e]) pairs
    float s[16];
#pragma unroll
    for (int e = 0; e < 16; ++e) s[e] = sbias;
    for (int c = 0; c < C; ++c) {
      float hqc = HQ[m * 65 + c];
      float wc = W[c];
#pragma unroll
      for (int e = 0; e < 16; ++e)
        s[e] += fmaxf(hqc + HKc[jl[e] * 64 + c], 0.f) * wc;
    }
#pragma unroll
    for (int e = 0; e < 16; ++e)
      s[e] = (jbase + jl[e] > ilocal) ? -__builtin_inff() : s[e] * 0.125f;

    // online softmax: row stats shared by lane pairs (L, L^16)
    float mx = -__builtin_inff();
#pragma unroll
    for (int e = 0; e < 16; ++e) mx = fmaxf(mx, s[e]);
    mx = fmaxf(mx, __shfl_xor(mx, 16, 32));
    float mnew = fmaxf(mrow, mx);
    float scale = __expf(mrow - mnew);

    v16bf ap;
    float ps = 0.f;
#pragma unroll
    for (int e = 0; e < 16; ++e) {
      float p = __expf(s[e] - mnew);
      ps += p;
      ap[e] = (__bf16)p;
    }
    lsum = lsum * scale + ps;
    mrow = mnew;

    // rescale accumulator rows (C-frag row of vgpr r is m = r + 8*(lane/16))
#pragma unroll
    for (int r = 0; r < 8; ++r) {
      float sc = __shfl(scale, r + kh, 32);
#pragma unroll
      for (int nc = 0; nc < 4; ++nc) acc[nc][r] *= sc;
    }

    // acc += P(16x32) @ V(32x64)
#pragma unroll
    for (int nc = 0; nc < 4; ++nc) {
      v16bf bv;
      int n = nc * 16 + m;
#pragma unroll
      for (int e = 0; e < 16; ++e) bv[e] = (__bf16)Vc[jl[e] * 64 + n];
      acc[nc] = wmma_bf16(ap, bv, acc[nc]);
    }
    __syncthreads();
  }

  float lfull = lsum + __shfl_xor(lsum, 16, 32);
#pragma unroll
  for (int r = 0; r < 8; ++r) {
    float lr = __shfl(lfull, r + kh, 32);
    float inv = 1.f / lr;
    int row = rowbase + r + kh;
#pragma unroll
    for (int nc = 0; nc < 4; ++nc)
      out[row * HS + nc * 16 + m] = acc[nc][r] * inv;
  }
}

extern "C" void kernel_launch(void* const* d_in, const int* in_sizes, int n_in,
                              void* d_out, int out_size, void* d_ws, size_t ws_size,
                              hipStream_t stream) {
  (void)in_sizes; (void)n_in; (void)out_size; (void)ws_size;
  const float* x     = (const float*)d_in[0];
  const float* pe_in = (const float*)d_in[1];
  const float* pw1   = (const float*)d_in[2];
  const float* pb1   = (const float*)d_in[3];
  const float* pw2   = (const float*)d_in[4];
  const float* pb2   = (const float*)d_in[5];
  const float* aw1   = (const float*)d_in[6];
  const float* ab1   = (const float*)d_in[7];
  const float* aw2   = (const float*)d_in[8];
  const float* ab2   = (const float*)d_in[9];
  const float* sw    = (const float*)d_in[10];
  const float* sb    = (const float*)d_in[11];
  const float* vw    = (const float*)d_in[12];

  char* ws = (char*)d_ws;
  __bf16* pw1b = (__bf16*)(ws + OFF_PW1B);
  __bf16* pw2b = (__bf16*)(ws + OFF_PW2B);
  __bf16* aw1b = (__bf16*)(ws + OFF_AW1B);
  __bf16* vwb  = (__bf16*)(ws + OFF_VWB);
  float* wvec  = (float*)(ws + OFF_WVEC);
  float* sbias = (float*)(ws + OFF_SBIAS);
  float* g_hk  = (float*)(ws + OFF_HK);
  float* g_hq  = (float*)(ws + OFF_HQ);
  float* g_v   = (float*)(ws + OFF_V);

  prep_kernel<<<1, 256, 0, stream>>>(pw1, pw2, aw1, vw, aw2, sw, ab2, sb, pw1b, pw2b,
                                     aw1b, vwb, wvec, sbias);
  proj_kernel<<<NROW / 16, 32, 0, stream>>>(x, pe_in, pb1, pb2, ab1, pw1b, pw2b, aw1b,
                                            vwb, g_hk, g_hq, g_v);
  attn_kernel<<<NROW / 16, 32, 0, stream>>>(g_hq, g_hk, g_v, wvec, sbias,
                                            (float*)d_out);
}
